// GraphAttn_87505663689363
// MI455X (gfx1250) — matrix operand order repaired and checked by reading, not compiled, not run on hardware
//
#include <hip/hip_runtime.h>

typedef __attribute__((ext_vector_type(16))) __bf16 v16bf;
typedef __attribute__((ext_vector_type(8)))  float  v8f;
typedef __attribute__((ext_vector_type(4)))  int    v4i;

union Frag { v16bf v; int4 q[2]; };

#define TKD 20  // dwords per padded LDS row (80 bytes: 64B data + 16B pad, 16B aligned)

#if defined(__gfx1250__) && __has_builtin(__builtin_amdgcn_global_load_async_to_lds_b128) && \
    __has_builtin(__builtin_amdgcn_s_wait_asynccnt)
#define USE_ASYNC_COPY 1
#else
#define USE_ASYNC_COPY 0
#endif

#if USE_ASYNC_COPY
__device__ __forceinline__ void async_copy_b128(const void* g, void* l)
{
    __builtin_amdgcn_global_load_async_to_lds_b128(
        (__attribute__((address_space(1))) v4i*)(g),
        (__attribute__((address_space(3))) v4i*)(l), 0, 0);
}
#endif

// ---------------------------------------------------------------------------
// Pack x / n : [b][ch][s] f32  ->  [b][s][ch] bf16   (LDS tiled transpose)
// ---------------------------------------------------------------------------
__global__ __launch_bounds__(256)
void pack_x_kernel(const float* __restrict__ x, const float* __restrict__ n,
                   __bf16* __restrict__ Xb, __bf16* __restrict__ Nb)
{
    __shared__ float tile[32][33];
    const int bx = blockIdx.x;            // 0..4095
    const int b  = bx >> 8;
    const int t  = bx & 255;
    const int cht = t >> 3;               // 0..31  (channel tile)
    const int st  = t & 7;                // 0..7   (spatial tile)
    const float* src = blockIdx.y ? n : x;
    __bf16*      dst = blockIdx.y ? Nb : Xb;
    const int tx = threadIdx.x & 31;
    const int ty = threadIdx.x >> 5;      // 0..7
#pragma unroll
    for (int q = 0; q < 4; ++q) {
        int ch = cht * 32 + ty + q * 8;
        int s  = st * 32 + tx;
        tile[ty + q * 8][tx] = src[((size_t)b << 18) + ((size_t)ch << 8) + s];
    }
    __syncthreads();
#pragma unroll
    for (int q = 0; q < 4; ++q) {
        int s  = st * 32 + ty + q * 8;
        int ch = cht * 32 + tx;
        dst[((size_t)b << 18) + ((size_t)s << 10) + ch] = (__bf16)tile[tx][ty + q * 8];
    }
}

// ---------------------------------------------------------------------------
// Pack W : [out][in][3][3] f32 -> Wb[tap][out][in] bf16
// ---------------------------------------------------------------------------
__global__ __launch_bounds__(128)
void pack_w_kernel(const float* __restrict__ W, __bf16* __restrict__ Wb)
{
    __shared__ float l[1152];
    const int bid = blockIdx.x;           // 0..8191
    const int no  = bid >> 3;
    const int ki0 = (bid & 7) << 7;
    const size_t base = (size_t)no * 9216 + (size_t)ki0 * 9;
    for (int idx = threadIdx.x; idx < 1152; idx += 128) l[idx] = W[base + idx];
    __syncthreads();
#pragma unroll
    for (int t = 0; t < 9; ++t)
        Wb[(size_t)t * 1048576 + (size_t)no * 1024 + ki0 + threadIdx.x] =
            (__bf16)l[threadIdx.x * 9 + t];
}

// ---------------------------------------------------------------------------
// Conv as implicit GEMM:  y[s, o] = sum_{tap,ki} Xb[b][shift(s,tap)][ki] * Wb[tap][o][ki]
// Tile 128x128, K-loop = 9 taps * 32 steps of K=32, double-buffered LDS with
// async global->LDS copies (ASYNCcnt) when available.
// Fused: +bias, h1t bf16 transposed store (z==0), s1/s2 = sum_s h*w_att (atomic).
// ---------------------------------------------------------------------------
__global__ __launch_bounds__(256)
void conv_wmma_kernel(const __bf16* __restrict__ Xb, const __bf16* __restrict__ Nb,
                      const __bf16* __restrict__ Wb, const float* __restrict__ bias,
                      const float* __restrict__ w1, const float* __restrict__ w2,
                      __bf16* __restrict__ h1t, float* __restrict__ s1,
                      float* __restrict__ s2)
{
    __shared__ int   As[2][128 * TKD];
    __shared__ int   Bs[2][128 * TKD];
    __shared__ float wl[128];

    const int tid  = threadIdx.x;
    const int lane = tid & 31;
    const int wave = tid >> 5;
    const int wm   = wave & 3;            // 32-row band
    const int wn   = wave >> 2;           // 64-col band

    const int bx = blockIdx.x;
    const int b  = bx >> 4;
    const int mt = (bx >> 3) & 1;
    const int nt = bx & 7;
    const int z  = blockIdx.y;            // 0: x-path (h1), 1: n-path (s2 only)

    const int sbase = mt * 128;
    const int nbase = nt * 128;

    const __bf16* A    = z ? Nb : Xb;
    const float*  wv   = z ? w2 : w1;
    float*        sdst = z ? s2 : s1;

    if (tid < 128) wl[tid] = wv[sbase + tid];

    v8f acc[2][4];
    const v8f vzero = {0.f, 0.f, 0.f, 0.f, 0.f, 0.f, 0.f, 0.f};
#pragma unroll
    for (int m = 0; m < 2; ++m)
#pragma unroll
        for (int nq = 0; nq < 4; ++nq) acc[m][nq] = vzero;

    const int arow  = tid >> 1;           // 0..127
    const int ahalf = tid & 1;
    const size_t Aoff_b = (size_t)b << 18;
    const int c  = (sbase + arow) >> 4;
    const int f  = (sbase + arow) & 15;
    const int ldst = arow * TKD + (ahalf << 3);

    auto gaddrA = [&](int kk) -> const int4* {
        const int t  = kk >> 5;
        const int k0 = (kk & 31) << 5;
        const int cc = (c + (t / 3) + 15) & 15;
        const int ff = (f + (t % 3) + 15) & 15;
        const int sp = (cc << 4) | ff;
        return (const int4*)(A + Aoff_b + ((size_t)sp << 10) + k0 + (ahalf << 4));
    };
    auto gaddrB = [&](int kk) -> const int4* {
        const int t  = kk >> 5;
        const int k0 = (kk & 31) << 5;
        return (const int4*)(Wb + ((size_t)t << 20) +
                             ((size_t)(nbase + arow) << 10) + k0 + (ahalf << 4));
    };

#if USE_ASYNC_COPY
    auto issue = [&](int kk, int buf) {
        const int4* ga = gaddrA(kk);
        const int4* gb = gaddrB(kk);
        async_copy_b128(ga,     &As[buf][ldst]);
        async_copy_b128(ga + 1, &As[buf][ldst + 4]);
        async_copy_b128(gb,     &Bs[buf][ldst]);
        async_copy_b128(gb + 1, &Bs[buf][ldst + 4]);
    };
    issue(0, 0);
    __builtin_amdgcn_s_wait_asynccnt(0);
    __syncthreads();
#else
    int4 ra0, ra1, rb0, rb1;
    {
        const int4* ga = gaddrA(0);
        const int4* gb = gaddrB(0);
        ra0 = ga[0]; ra1 = ga[1]; rb0 = gb[0]; rb1 = gb[1];
        *(int4*)&As[0][ldst] = ra0; *(int4*)&As[0][ldst + 4] = ra1;
        *(int4*)&Bs[0][ldst] = rb0; *(int4*)&Bs[0][ldst + 4] = rb1;
    }
    __syncthreads();
#endif

    const int hsel = (lane < 16) ? 0 : 1;

    for (int kk = 0; kk < 288; ++kk) {
        const int cur = kk & 1;

#if USE_ASYNC_COPY
        if (kk + 1 < 288) issue(kk + 1, cur ^ 1);
#else
        int4 ra0n, ra1n, rb0n, rb1n;
        if (kk + 1 < 288) {
            const int4* ga = gaddrA(kk + 1);
            const int4* gb = gaddrB(kk + 1);
            ra0n = ga[0]; ra1n = ga[1]; rb0n = gb[0]; rb1n = gb[1];
        }
#endif

        // fragments from current buffer
        Frag af[2];
#pragma unroll
        for (int ms = 0; ms < 2; ++ms) {
            const int r = wm * 32 + ms * 16 + (lane & 15);
            const int* p = &As[cur][r * TKD];
            af[ms].q[0] = *(const int4*)(p + hsel * 4);       // K 0-7 / 8-15
            af[ms].q[1] = *(const int4*)(p + 8 + hsel * 4);   // K 16-23 / 24-31
        }
        Frag fb[4];
#pragma unroll
        for (int ns = 0; ns < 4; ++ns) {
            const int r = wn * 64 + ns * 16 + (lane & 15);
            const int* p = &Bs[cur][r * TKD];
            fb[ns].q[0] = *(const int4*)(p + hsel * 4);       // K 0-7 / 8-15
            fb[ns].q[1] = *(const int4*)(p + 8 + hsel * 4);
        }
#pragma unroll
        for (int ms = 0; ms < 2; ++ms)
#pragma unroll
            for (int ns = 0; ns < 4; ++ns)
                acc[ms][ns] = __builtin_amdgcn_wmma_f32_16x16x32_bf16(
                    false, af[ms].v, false, fb[ns].v, (short)0, acc[ms][ns], false, false);

#if USE_ASYNC_COPY
        __builtin_amdgcn_s_wait_asynccnt(0);
#else
        if (kk + 1 < 288) {
            const int nb_ = cur ^ 1;
            *(int4*)&As[nb_][ldst] = ra0n; *(int4*)&As[nb_][ldst + 4] = ra1n;
            *(int4*)&Bs[nb_][ldst] = rb0n; *(int4*)&Bs[nb_][ldst + 4] = rb1n;
        }
#endif
        __syncthreads();
    }

    // epilogue: bias, attention-weight dot (-> s1/s2), transposed bf16 store of h1
    const int col  = lane & 15;
    const int rofs = (lane >> 4) << 3;    // rows r / r+8
#pragma unroll
    for (int ns = 0; ns < 4; ++ns) {
        const int nch = nbase + wn * 64 + ns * 16 + col;
        const float bv = bias[nch];
        float part = 0.f;
#pragma unroll
        for (int ms = 0; ms < 2; ++ms) {
            const int sl = wm * 32 + ms * 16 + rofs;
#pragma unroll
            for (int r = 0; r < 8; ++r) {
                const float hv = acc[ms][ns][r] + bv;
                part += hv * wl[sl + r];
                if (z == 0) {
                    const size_t idx = (((size_t)b * 256 + sbase + sl + r) << 10) + nch;
                    h1t[idx] = (__bf16)hv;
                }
            }
        }
        atomicAdd(&sdst[b * 1024 + nch], part);
    }
}

// ---------------------------------------------------------------------------
// Column softmax stats: cb[b,j] = -M_j - ln( sum_i exp(lrelu(s1_i+s2_j) - M_j) )
// M_j = lrelu(max_i s1_i + s2_j)  (lrelu is monotone)
// ---------------------------------------------------------------------------
__global__ __launch_bounds__(256)
void scores_kernel(const float* __restrict__ s1, const float* __restrict__ s2,
                   float* __restrict__ cb)
{
    __shared__ float s1l[1024];
    __shared__ float red[256];
    const int b = blockIdx.x, tid = threadIdx.x;
    float mx = -1e30f;
#pragma unroll
    for (int q = 0; q < 4; ++q) {
        const float v = s1[b * 1024 + tid + q * 256];
        s1l[tid + q * 256] = v;
        mx = fmaxf(mx, v);
    }
    red[tid] = mx;
    __syncthreads();
    for (int w = 128; w > 0; w >>= 1) {
        if (tid < w) red[tid] = fmaxf(red[tid], red[tid + w]);
        __syncthreads();
    }
    const float m1 = red[0];
#pragma unroll
    for (int q = 0; q < 4; ++q) {
        const int j = tid + q * 256;
        const float s2j = s2[b * 1024 + j];
        const float t0 = m1 + s2j;
        const float M = fmaxf(t0, 0.2f * t0);
        float d = 0.f;
        for (int i = 0; i < 1024; ++i) {
            const float u = s1l[i] + s2j;
            d += __expf(fmaxf(u, 0.2f * u) - M);
        }
        cb[b * 1024 + j] = -M - __logf(d);
    }
}

// ---------------------------------------------------------------------------
// out[b,i,s] = sum_j exp(lrelu(s1_i+s2_j)+cb_j) * h1[b,j,s]
// A (attention) built in-register per WMMA fragment layout; B = h1t tiles.
// ---------------------------------------------------------------------------
__global__ __launch_bounds__(256)
void attn_wmma_kernel(const float* __restrict__ s1, const float* __restrict__ s2,
                      const float* __restrict__ cb, const __bf16* __restrict__ h1t,
                      float* __restrict__ out)
{
    __shared__ int Bs[128 * TKD];
    __shared__ __align__(16) float s2c[32];
    __shared__ __align__(16) float cbc[32];

    const int tid  = threadIdx.x;
    const int lane = tid & 31;
    const int wave = tid >> 5;
    const int wm   = wave & 3;
    const int wn   = wave >> 2;

    const int bx = blockIdx.x;
    const int b  = bx >> 4;
    const int mt = (bx >> 1) & 7;
    const int nt = bx & 1;
    const int i0  = mt * 128;
    const int sp0 = nt * 128;

    float s1v[2];
#pragma unroll
    for (int ms = 0; ms < 2; ++ms)
        s1v[ms] = s1[b * 1024 + i0 + wm * 32 + ms * 16 + (lane & 15)];

    v8f acc[2][4];
    const v8f vzero = {0.f, 0.f, 0.f, 0.f, 0.f, 0.f, 0.f, 0.f};
#pragma unroll
    for (int m = 0; m < 2; ++m)
#pragma unroll
        for (int nq = 0; nq < 4; ++nq) acc[m][nq] = vzero;

    const int col  = tid >> 1;
    const int half = tid & 1;

    for (int kk = 0; kk < 32; ++kk) {
        const int j0 = kk << 5;
        const int4* bg = (const int4*)(h1t + (((size_t)b * 256 + sp0 + col) << 10) +
                                       j0 + (half << 4));
        int4 b0 = bg[0], b1 = bg[1];
        if (kk + 1 < 32)
            __builtin_prefetch(h1t + (((size_t)b * 256 + sp0 + col) << 10) +
                               ((kk + 1) << 5) + (half << 4), 0, 1);
        const int dst = col * TKD + (half << 3);
        *(int4*)&Bs[dst] = b0; *(int4*)&Bs[dst + 4] = b1;
        if (tid < 32)       s2c[tid]      = s2[b * 1024 + j0 + tid];
        else if (tid < 64)  cbc[tid - 32] = cb[b * 1024 + j0 + tid - 32];
        __syncthreads();

        const int off  = (lane < 16) ? 0 : 8;
        const int hsel = (lane < 16) ? 0 : 1;
        float s2r[16], cbr[16];
        *(float4*)&s2r[0]  = *(const float4*)&s2c[off];
        *(float4*)&s2r[4]  = *(const float4*)&s2c[off + 4];
        *(float4*)&s2r[8]  = *(const float4*)&s2c[off + 16];
        *(float4*)&s2r[12] = *(const float4*)&s2c[off + 20];
        *(float4*)&cbr[0]  = *(const float4*)&cbc[off];
        *(float4*)&cbr[4]  = *(const float4*)&cbc[off + 4];
        *(float4*)&cbr[8]  = *(const float4*)&cbc[off + 16];
        *(float4*)&cbr[12] = *(const float4*)&cbc[off + 20];

        Frag fb[4];
#pragma unroll
        for (int ns = 0; ns < 4; ++ns) {
            const int r = wn * 64 + ns * 16 + (lane & 15);
            const int* p = &Bs[r * TKD];
            fb[ns].q[0] = *(const int4*)(p + hsel * 4);
            fb[ns].q[1] = *(const int4*)(p + 8 + hsel * 4);
        }
#pragma unroll
        for (int ms = 0; ms < 2; ++ms) {
            Frag af;
#pragma unroll
            for (int e = 0; e < 16; ++e) {
                const float tt = s1v[ms] + s2r[e];
                const float lr = fmaxf(tt, 0.2f * tt);
                af.v[e] = (__bf16)__expf(lr + cbr[e]);
            }
#pragma unroll
            for (int ns = 0; ns < 4; ++ns)
                acc[ms][ns] = __builtin_amdgcn_wmma_f32_16x16x32_bf16(
                    false, af.v, false, fb[ns].v, (short)0, acc[ms][ns], false, false);
        }
        __syncthreads();
    }

    const int colx = lane & 15;
    const int rofs = (lane >> 4) << 3;
#pragma unroll
    for (int ms = 0; ms < 2; ++ms)
#pragma unroll
        for (int ns = 0; ns < 4; ++ns)
#pragma unroll
            for (int r = 0; r < 8; ++r) {
                const int i    = i0 + wm * 32 + ms * 16 + rofs + r;
                const int sidx = sp0 + wn * 64 + ns * 16 + colx;
                out[(((size_t)b << 10) + i) * 256 + sidx] = acc[ms][ns][r];
            }
}

// ---------------------------------------------------------------------------
extern "C" void kernel_launch(void* const* d_in, const int* in_sizes, int n_in,
                              void* d_out, int out_size, void* d_ws, size_t ws_size,
                              hipStream_t stream)
{
    const float* x    = (const float*)d_in[0];
    const float* nn   = (const float*)d_in[1];
    const float* W    = (const float*)d_in[2];
    const float* bias = (const float*)d_in[3];
    const float* w1   = (const float*)d_in[4];
    const float* w2   = (const float*)d_in[5];

    char* ws = (char*)d_ws;
    __bf16* Xb  = (__bf16*)(ws);                 //  8,388,608 B
    __bf16* Nb  = (__bf16*)(ws + 8388608);       //  8,388,608 B
    __bf16* Wb  = (__bf16*)(ws + 16777216);      // 18,874,368 B
    __bf16* h1t = (__bf16*)(ws + 35651584);      //  8,388,608 B
    float*  s1  = (float*)(ws + 44040192);       //     65,536 B
    float*  s2  = (float*)(ws + 44105728);       //     65,536 B
    float*  cb  = (float*)(ws + 44171264);       //     65,536 B
    float*  out = (float*)d_out;

    (void)hipMemsetAsync(s1, 0, 131072, stream); // zero s1+s2 (contiguous)

    pack_x_kernel<<<dim3(4096, 2), 256, 0, stream>>>(x, nn, Xb, Nb);
    pack_w_kernel<<<8192, 128, 0, stream>>>(W, Wb);
    conv_wmma_kernel<<<dim3(256, 2), 256, 0, stream>>>(Xb, Nb, Wb, bias, w1, w2,
                                                       h1t, s1, s2);
    scores_kernel<<<16, 256, 0, stream>>>(s1, s2, cb);
    attn_wmma_kernel<<<256, 256, 0, stream>>>(s1, s2, cb, h1t, out);
}
